// CausalSelfAttention_2551210574649
// MI455X (gfx1250) — compile-verified, matrix operand order checked by reading, not run on hardware
//
#include <hip/hip_runtime.h>
#include <hip/hip_bf16.h>

typedef __bf16 bf16;
typedef __attribute__((ext_vector_type(16))) __bf16 v16bf;
typedef __attribute__((ext_vector_type(8)))  __bf16 v8bf;
typedef __attribute__((ext_vector_type(8)))  float  v8f;
typedef __attribute__((ext_vector_type(4)))  float  v4f;

#define B_  2
#define T_  4096
#define C_  768
#define H_  12
#define D_  64

// Build a 16-halfword A/B WMMA fragment from two contiguous 16-byte chunks.
__device__ __forceinline__ v16bf frag16(const bf16* p0, const bf16* p1) {
  v8bf a = *(const v8bf*)p0;
  v8bf b = *(const v8bf*)p1;
  return __builtin_shufflevector(a, b, 0,1,2,3,4,5,6,7,8,9,10,11,12,13,14,15);
}

// Vectorized f32 -> bf16 cast: 8 elements/thread (2x float4 in, 16B out).
__global__ __launch_bounds__(256) void cast_f32_bf16(const float* __restrict__ s,
                                                     bf16* __restrict__ d, int n8) {
  int i = blockIdx.x * 256 + threadIdx.x;
  if (i >= n8) return;
  v4f f0 = *(const v4f*)(s + (size_t)i * 8);
  v4f f1 = *(const v4f*)(s + (size_t)i * 8 + 4);
  v8bf o;
  #pragma unroll
  for (int e = 0; e < 4; ++e) { o[e] = (bf16)f0[e]; o[e + 4] = (bf16)f1[e]; }
  *(v8bf*)(d + (size_t)i * 8) = o;
}

// ---------------------------------------------------------------------------
// Double-buffered bf16 GEMM: Cacc(f32) = A[M,K] * W[K,N] + bias[N]
// MODE 0: scatter bf16 into q/k/v [B,H,T,D] buffers (N = 3C)
// MODE 1: write f32 out[M,N] (N = C)
// Block 256 thr = 8 waves, tile 128(M) x 256(N) x 32(K).
// Wave grid 2(M) x 4(N); each wave: 4 m-subtiles x 4 n-subtiles of 16x16
// -> 16 WMMAs per k-step against 16 ds_load_b128 fragment loads (1:1).
// Global loads of tile k+1 are issued before the WMMAs of tile k.
// ---------------------------------------------------------------------------
template<int MODE>
__global__ __launch_bounds__(256) void gemm_bf16(
    const bf16* __restrict__ A, const bf16* __restrict__ W,
    const float* __restrict__ bias,
    bf16* __restrict__ q, bf16* __restrict__ k, bf16* __restrict__ v,
    float* __restrict__ outp, int N)
{
  const int K = C_;
  __shared__ bf16 aS[2][128 * 40];  // [m][k], stride 40 hw (80B, 16B-aligned)
  __shared__ bf16 bS[2][256 * 40];  // [n][k] (transposed weight tile)

  const int tid  = threadIdx.x;
  const int lane = tid & 31, wid = tid >> 5;
  const int wm = wid >> 2, wn = wid & 3;
  const int hi = lane >> 4, lm = lane & 15;
  const int M0 = blockIdx.y * 128, N0 = blockIdx.x * 256;

  const int rA = tid >> 1, halfA = tid & 1;          // A staging map (16 elems)
  const int kkB = tid >> 3, n0B = (tid & 7) * 32;    // B staging map (32 elems)

  v8bf a0, a1, b0, b1, b2, b3;
  auto loadG = [&](int k0) {
    const bf16* srcA = A + (size_t)(M0 + rA) * K + k0 + halfA * 16;
    a0 = *(const v8bf*)(srcA);
    a1 = *(const v8bf*)(srcA + 8);
    const bf16* srcB = W + (size_t)(k0 + kkB) * N + N0 + n0B;
    b0 = *(const v8bf*)(srcB);
    b1 = *(const v8bf*)(srcB + 8);
    b2 = *(const v8bf*)(srcB + 16);
    b3 = *(const v8bf*)(srcB + 24);
  };
  auto storeL = [&](int p) {
    bf16* dst = aS[p] + rA * 40 + halfA * 16;
    *(v8bf*)(dst)     = a0;
    *(v8bf*)(dst + 8) = a1;
    #pragma unroll
    for (int e = 0; e < 8; ++e) {
      bS[p][(n0B + e) * 40 + kkB]      = b0[e];
      bS[p][(n0B + 8 + e) * 40 + kkB]  = b1[e];
      bS[p][(n0B + 16 + e) * 40 + kkB] = b2[e];
      bS[p][(n0B + 24 + e) * 40 + kkB] = b3[e];
    }
  };

  v8f acc[4][4];
  #pragma unroll
  for (int i = 0; i < 4; ++i)
    #pragma unroll
    for (int j = 0; j < 4; ++j)
      #pragma unroll
      for (int e = 0; e < 8; ++e) acc[i][j][e] = 0.0f;

  loadG(0);
  storeL(0);
  int p = 0;
  for (int k0 = 0; k0 < K; k0 += 32) {
    __syncthreads();                 // buffer p staged by all waves
    const bool more = (k0 + 32 < K);
    if (more) loadG(k0 + 32);        // overlap next-tile loads with WMMAs

    v16bf aF[4], bF[4];
    #pragma unroll
    for (int i = 0; i < 4; ++i) {
      int m = wm * 64 + i * 16 + lm;
      aF[i] = frag16(&aS[p][m * 40 + hi * 8], &aS[p][m * 40 + 16 + hi * 8]);
    }
    #pragma unroll
    for (int j = 0; j < 4; ++j) {
      int n = wn * 64 + j * 16 + lm;
      bF[j] = frag16(&bS[p][n * 40 + hi * 8], &bS[p][n * 40 + 16 + hi * 8]);
    }
    #pragma unroll
    for (int i = 0; i < 4; ++i)
      #pragma unroll
      for (int j = 0; j < 4; ++j)
        acc[i][j] = __builtin_amdgcn_wmma_f32_16x16x32_bf16(
            false, aF[i], false, bF[j], (short)0, acc[i][j], false, false);

    if (more) storeL(p ^ 1);
    p ^= 1;
  }

  // Epilogue: C/D layout -> element (m = r + 8*hi, n = lane%16)
  #pragma unroll
  for (int i = 0; i < 4; ++i) {
    #pragma unroll
    for (int j = 0; j < 4; ++j) {
      #pragma unroll
      for (int r = 0; r < 8; ++r) {
        int m = M0 + wm * 64 + i * 16 + r + 8 * hi;
        int n = N0 + wn * 64 + j * 16 + lm;
        float val = acc[i][j][r] + bias[n];
        if (MODE == 0) {
          int which = n / C_;
          int c = n % C_;
          int hh = c >> 6, d = c & 63;
          int bb = m >> 12, t = m & (T_ - 1);
          bf16* dst = (which == 0) ? q : (which == 1) ? k : v;
          dst[(((size_t)bb * H_ + hh) * T_ + t) * D_ + d] = (bf16)val;
        } else {
          outp[(size_t)m * C_ + n] = val;
        }
      }
    }
  }
}

// ---------------------------------------------------------------------------
// Flash attention: grid (T/128, B*H), block 256 thr = 8 waves.
// Each wave owns 16 q-rows (128-row q-tile); 64-key K/V tiles staged in LDS.
// Wave-uniform causal skip for fully-masked tiles; per-element mask only on
// diagonal tiles. Next K/V tile prefetched (global_prefetch_b8).
// ---------------------------------------------------------------------------
__global__ __launch_bounds__(256) void flash_attn(
    const bf16* __restrict__ q, const bf16* __restrict__ k,
    const bf16* __restrict__ v, bf16* __restrict__ y)
{
  __shared__ bf16 kS[64 * 72];       // [key][d], stride 72 hw = 144B
  __shared__ bf16 vT[64 * 72];       // [d][key] (transposed V tile)
  __shared__ bf16 pS[8 * 16 * 72];   // per-wave P strip for C->A layout fix

  const int tid  = threadIdx.x;
  const int lane = tid & 31, wid = tid >> 5;
  const int hi = lane >> 4, lm = lane & 15;
  const int bh = blockIdx.y;          // b*H + h
  const int qt = blockIdx.x;
  const size_t head = (size_t)bh * T_ * D_;
  const int row0 = qt * 128 + wid * 16;   // this wave's first q row

  // Q fragments for this wave's 16 rows, K=D=64 split into 2 WMMA k-steps
  v16bf qf[2];
  {
    const bf16* qrow = q + head + (size_t)(row0 + lm) * D_;
    #pragma unroll
    for (int ks = 0; ks < 2; ++ks)
      qf[ks] = frag16(qrow + ks * 32 + hi * 8, qrow + ks * 32 + 16 + hi * 8);
  }

  v8f o[4];
  #pragma unroll
  for (int j = 0; j < 4; ++j)
    #pragma unroll
    for (int e = 0; e < 8; ++e) o[j][e] = 0.0f;
  float rmax[8], rsum[8];
  #pragma unroll
  for (int r = 0; r < 8; ++r) { rmax[r] = -1e30f; rsum[r] = 0.0f; }

  const int rS = tid >> 2, q4 = tid & 3;  // staging map: row 0..63, 16-col quarter
  const int ktmax = 2 * qt + 1;           // last k-tile touching this q-tile

  for (int kt = 0; kt <= ktmax; ++kt) {
    __syncthreads();
    { // stage K (row-major) and V (transposed) tiles; prefetch next tiles
      const bf16* ksrc = k + head + (size_t)(kt * 64 + rS) * D_ + q4 * 16;
      const bf16* vsrc = v + head + (size_t)(kt * 64 + rS) * D_ + q4 * 16;
      v8bf tk0 = *(const v8bf*)(ksrc);
      v8bf tk1 = *(const v8bf*)(ksrc + 8);
      *(v8bf*)&kS[rS * 72 + q4 * 16]     = tk0;
      *(v8bf*)&kS[rS * 72 + q4 * 16 + 8] = tk1;
      v8bf tv0 = *(const v8bf*)(vsrc);
      v8bf tv1 = *(const v8bf*)(vsrc + 8);
      #pragma unroll
      for (int e = 0; e < 8; ++e) {
        vT[(q4 * 16 + e) * 72 + rS]     = tv0[e];
        vT[(q4 * 16 + 8 + e) * 72 + rS] = tv1[e];
      }
      if (kt < ktmax) {
        __builtin_prefetch(ksrc + 64 * D_, 0, 3);
        __builtin_prefetch(vsrc + 64 * D_, 0, 3);
      }
    }
    __syncthreads();

    // Wave-uniform causal skip: this wave sees no valid keys in this tile.
    if (kt * 64 > row0 + 15) continue;
    const bool diag = (kt * 64 + 63 > row0);   // tile needs per-element mask

    // S = Q * K^T  (row-major K rows == B-operand layout: lane = key, hw = d)
    v8f s[4];
    #pragma unroll
    for (int j = 0; j < 4; ++j)
      #pragma unroll
      for (int e = 0; e < 8; ++e) s[j][e] = 0.0f;
    #pragma unroll
    for (int j = 0; j < 4; ++j) {
      int n = j * 16 + lm;
      #pragma unroll
      for (int ks = 0; ks < 2; ++ks) {
        v16bf bF = frag16(&kS[n * 72 + ks * 32 + hi * 8],
                          &kS[n * 72 + ks * 32 + 16 + hi * 8]);
        s[j] = __builtin_amdgcn_wmma_f32_16x16x32_bf16(
            false, qf[ks], false, bF, (short)0, s[j], false, false);
      }
    }

    // scale + causal mask (diagonal tiles only)
    const float sc = 0.125f;   // 1/sqrt(64)
    #pragma unroll
    for (int j = 0; j < 4; ++j)
      #pragma unroll
      for (int r = 0; r < 8; ++r) {
        float val = s[j][r] * sc;
        if (diag) {
          int tq = row0 + r + 8 * hi;
          int tk = kt * 64 + j * 16 + lm;
          if (tk > tq) val = -1e30f;
        }
        s[j][r] = val;
      }

    // online softmax; a row lives on the 16 lanes sharing (r, hi)
    float f8[8];
    #pragma unroll
    for (int r = 0; r < 8; ++r) {
      float mx = fmaxf(fmaxf(s[0][r], s[1][r]), fmaxf(s[2][r], s[3][r]));
      mx = fmaxf(mx, __shfl_xor(mx, 1));
      mx = fmaxf(mx, __shfl_xor(mx, 2));
      mx = fmaxf(mx, __shfl_xor(mx, 4));
      mx = fmaxf(mx, __shfl_xor(mx, 8));
      float nm = fmaxf(rmax[r], mx);
      f8[r] = __expf(rmax[r] - nm);
      rmax[r] = nm;
      float sm = 0.0f;
      #pragma unroll
      for (int j = 0; j < 4; ++j) {
        float pv = __expf(s[j][r] - nm);
        s[j][r] = pv;
        sm += pv;
      }
      sm += __shfl_xor(sm, 1);
      sm += __shfl_xor(sm, 2);
      sm += __shfl_xor(sm, 4);
      sm += __shfl_xor(sm, 8);
      rsum[r] = rsum[r] * f8[r] + sm;
    }
    #pragma unroll
    for (int j = 0; j < 4; ++j)
      #pragma unroll
      for (int r = 0; r < 8; ++r) o[j][r] *= f8[r];

    // P: C-layout -> per-wave LDS strip -> A-layout fragments
    bf16* pw = pS + wid * 16 * 72;
    #pragma unroll
    for (int j = 0; j < 4; ++j)
      #pragma unroll
      for (int r = 0; r < 8; ++r)
        pw[(r + 8 * hi) * 72 + j * 16 + lm] = (bf16)s[j][r];
    v16bf pf[2];
    #pragma unroll
    for (int ks = 0; ks < 2; ++ks)
      pf[ks] = frag16(&pw[lm * 72 + ks * 32 + hi * 8],
                      &pw[lm * 72 + ks * 32 + 16 + hi * 8]);

    // O += P * V  (B operand from transposed V: lane = d, hw = key)
    #pragma unroll
    for (int j = 0; j < 4; ++j) {
      int d = j * 16 + lm;
      #pragma unroll
      for (int ks = 0; ks < 2; ++ks) {
        v16bf bF = frag16(&vT[d * 72 + ks * 32 + hi * 8],
                          &vT[d * 72 + ks * 32 + 16 + hi * 8]);
        o[j] = __builtin_amdgcn_wmma_f32_16x16x32_bf16(
            false, pf[ks], false, bF, (short)0, o[j], false, false);
      }
    }
  }

  // epilogue: y[b, t, h*64+d] bf16
  int b = bh / H_, h = bh % H_;
  #pragma unroll
  for (int j = 0; j < 4; ++j)
    #pragma unroll
    for (int r = 0; r < 8; ++r) {
      int tq  = row0 + r + 8 * hi;
      int col = h * 64 + j * 16 + lm;
      y[((size_t)b * T_ + tq) * C_ + col] = (bf16)(o[j][r] / rsum[r]);
    }
}

extern "C" void kernel_launch(void* const* d_in, const int* in_sizes, int n_in,
                              void* d_out, int out_size, void* d_ws, size_t ws_size,
                              hipStream_t stream) {
  (void)in_sizes; (void)n_in; (void)out_size; (void)ws_size;
  const float* x      = (const float*)d_in[0];
  const float* w_attn = (const float*)d_in[1];
  const float* b_attn = (const float*)d_in[2];
  const float* w_proj = (const float*)d_in[3];
  const float* b_proj = (const float*)d_in[4];
  float* out = (float*)d_out;

  char* ws = (char*)d_ws;
  size_t off = 0;
  auto alloc = [&](size_t bytes) -> char* {
    char* p = ws + off;
    off += (bytes + 255) & ~(size_t)255;
    return p;
  };
  const size_t nx  = (size_t)B_ * T_ * C_;
  const size_t nwa = (size_t)C_ * 3 * C_;
  const size_t nwp = (size_t)C_ * C_;
  const size_t nh  = (size_t)B_ * H_ * T_ * D_;

  bf16* xb  = (bf16*)alloc(nx * 2);
  bf16* wab = (bf16*)alloc(nwa * 2);
  bf16* wpb = (bf16*)alloc(nwp * 2);
  bf16* qb  = (bf16*)alloc(nh * 2);
  bf16* kb  = (bf16*)alloc(nh * 2);
  bf16* vb  = (bf16*)alloc(nh * 2);
  bf16* yb  = (bf16*)alloc(nx * 2);

  cast_f32_bf16<<<(int)((nx / 8 + 255) / 256), 256, 0, stream>>>(x, xb, (int)(nx / 8));
  cast_f32_bf16<<<(int)((nwa / 8 + 255) / 256), 256, 0, stream>>>(w_attn, wab, (int)(nwa / 8));
  cast_f32_bf16<<<(int)((nwp / 8 + 255) / 256), 256, 0, stream>>>(w_proj, wpb, (int)(nwp / 8));

  gemm_bf16<0><<<dim3(3 * C_ / 256, B_ * T_ / 128), 256, 0, stream>>>(
      xb, wab, b_attn, qb, kb, vb, nullptr, 3 * C_);

  flash_attn<<<dim3(T_ / 128, B_ * H_), 256, 0, stream>>>(qb, kb, vb, yb);

  gemm_bf16<1><<<dim3(C_ / 256, B_ * T_ / 128), 256, 0, stream>>>(
      yb, wpb, b_proj, nullptr, nullptr, nullptr, out, C_);
}
